// BiLSTM_CRF_26912265077326
// MI455X (gfx1250) — matrix operand order repaired and checked by reading, not compiled, not run on hardware
//
#include <hip/hip_runtime.h>
#include <hip/hip_bf16.h>
#include <stdint.h>

// ---------------- problem dimensions ----------------
#define Vv   32000
#define Tt   48
#define Em   128      // E
#define Hm   128      // H (per direction)
#define G4   512      // 4*H
#define HID2 256      // 2*H
#define Bsz  256      // B
#define Lm   512      // L

typedef _Float16 f16;
typedef __attribute__((ext_vector_type(16))) _Float16 v16h;
typedef __attribute__((ext_vector_type(8)))  _Float16 v8h;
typedef __attribute__((ext_vector_type(4)))  _Float16 v4h;
typedef __attribute__((ext_vector_type(8)))  float    v8f;

// ---- WMMA fragment loaders (wave32; this file's contiguous k order, A and B alike) ----
// A (16x32 f16): lane L -> row M = L%16; halves0-15 = K[k0+kbo .. +16)
// with kbo = (L>=16)*16.
__device__ __forceinline__ v16h ldA16(const f16* p) {
  v8h lo = *(const v8h*)(p);
  v8h hi = *(const v8h*)(p + 8);
  return __builtin_shufflevector(lo, hi, 0,1,2,3,4,5,6,7,8,9,10,11,12,13,14,15);
}
// B (32x16 f16): lane L -> col N = L%16; halves0-15 = K[k0+(L>=16)*16 .. +16)
// -> contiguous from row-major W[col][k].

__device__ __forceinline__ float sigmoidf_(float x) { return 1.0f / (1.0f + __expf(-x)); }

// ---------------- 1) embedding gather -> f16, time-major (L*B, E) ----------------
__global__ void k_embed(const int* __restrict__ words, const float* __restrict__ emb,
                        f16* __restrict__ x16) {
  const size_t N = (size_t)Lm * Bsz * Em;
  for (size_t i = (size_t)blockIdx.x * blockDim.x + threadIdx.x; i < N;
       i += (size_t)gridDim.x * blockDim.x) {
    size_t r = i >> 7;              // row = l*B + b
    int e = (int)(i & 127);
    int l = (int)(r >> 8);          // / Bsz
    int b = (int)(r & 255);
    int w = words[(size_t)b * Lm + l];   // padding rows are 0 -> emb row 0 is zero
    x16[i] = (f16)emb[(size_t)w * Em + e];
  }
}

// ---------------- 2) Wout -> f16 ----------------
__global__ void k_cvt_wout(const float* __restrict__ Wout, f16* __restrict__ Wout16) {
  for (int i = threadIdx.x; i < Tt * HID2; i += 256) Wout16[i] = (f16)Wout[i];
}

// ---------------- 3) recurrent BiLSTM kernel, register-resident weights ----------
// grid = (B/16, 2 dirs), 512 threads = 16 waves; wave wv owns gate cols [wv*32, wv*32+32).
// Weight B-fragments (2 nt x 4 kt x 2 mats = 16 x v16h = 128 VGPRs) live in registers
// for the whole time loop; x A-fragments are software-pipelined one step ahead and
// prefetched (WGP scope) two steps ahead.
__global__ void __launch_bounds__(512)
k_lstm(const f16* __restrict__ x16,          // (L*B, E)
       const float* __restrict__ Wih_f, const float* __restrict__ Whh_f,
       const float* __restrict__ bih_f, const float* __restrict__ bhh_f,
       const float* __restrict__ Wih_b, const float* __restrict__ Whh_b,
       const float* __restrict__ bih_b, const float* __restrict__ bhh_b,
       f16* __restrict__ h_all) {            // (L*B, 2H)
  __shared__ float sGate[16 * G4];           // 32 KB gate-exchange tile
  __shared__ f16   sH[16 * Hm];              // 4 KB h_{t-1} tile

  const int dir = blockIdx.y;
  const float* __restrict__ Wih = dir ? Wih_b : Wih_f;
  const float* __restrict__ Whh = dir ? Whh_b : Whh_f;
  const float* __restrict__ bih = dir ? bih_b : bih_f;
  const float* __restrict__ bhh = dir ? bhh_b : bhh_f;

  const int tid  = threadIdx.x;              // 0..511
  const int lane = tid & 31;
  const int wv   = tid >> 5;                 // 0..15
  const int b0   = blockIdx.x * 16;
  const int ncol = lane & 15;
  const int arow = lane & 15;
  const int koff = (lane >> 4) << 3;         // unused; A frags use kbo
  const int kbo  = (lane >> 4) << 4;         // B-frag half offset
  const int mrow = (lane >> 4) * 8;          // C/D frag row base

  // ---- hoist weight fragments into registers (f32 -> f16 convert, once) ----
  v16h wih[2][4], whh[2][4];
  float bias[2];
#pragma unroll
  for (int nt = 0; nt < 2; ++nt) {
    const int col = wv * 32 + nt * 16 + ncol;
    bias[nt] = bih[col] + bhh[col];
#pragma unroll
    for (int kt = 0; kt < 4; ++kt) {
      const float* pi = Wih + (size_t)col * Em + kt * 32 + kbo;
      const float* ph = Whh + (size_t)col * Hm + kt * 32 + kbo;
      v16h a, b;
#pragma unroll
      for (int q = 0; q < 4; ++q) {
        float4 fi = *(const float4*)(pi + q * 4);
        float4 fh = *(const float4*)(ph + q * 4);
        a[q*4+0] = (f16)fi.x; a[q*4+1] = (f16)fi.y; a[q*4+2] = (f16)fi.z; a[q*4+3] = (f16)fi.w;
        b[q*4+0] = (f16)fh.x; b[q*4+1] = (f16)fh.y; b[q*4+2] = (f16)fh.z; b[q*4+3] = (f16)fh.w;
      }
      wih[nt][kt] = a; whh[nt][kt] = b;
    }
  }

  for (int i = tid; i < 16 * Hm; i += 512) sH[i] = (f16)0.0f;
  __syncthreads();

  // cell ownership: thread tid -> row cm = tid/32, cols [4*lane, 4*lane+4)
  const int cm = wv;
  const int cn = lane * 4;
  float c[4] = {0.f, 0.f, 0.f, 0.f};

  // software pipeline: axn holds step s+1's x fragments
  v16h axn[4];
  {
    const int l0 = dir ? (Lm - 1) : 0;
    const f16* xr = x16 + (((size_t)l0 * Bsz + b0 + arow)) * Em;
#pragma unroll
    for (int kt = 0; kt < 4; ++kt) axn[kt] = ldA16(xr + kt * 32 + kbo);
  }

  for (int s = 0; s < Lm; ++s) {
    const int l = dir ? (Lm - 1 - s) : s;
    const size_t rowbase = (size_t)l * Bsz + b0;

    // consume pipelined fragments; immediately issue next step's loads
    v16h ax[4];
#pragma unroll
    for (int kt = 0; kt < 4; ++kt) ax[kt] = axn[kt];
    {
      const int ln  = dir ? (l > 0 ? l - 1 : 0) : (l < Lm - 1 ? l + 1 : l);
      const int lnn = dir ? (ln > 0 ? ln - 1 : 0) : (ln < Lm - 1 ? ln + 1 : ln);
      const f16* xr = x16 + ((size_t)ln * Bsz + b0 + arow) * Em;
      // prefetch 2 steps ahead into WGP cache (locality 3 -> WGP scope)
      __builtin_prefetch(x16 + ((size_t)lnn * Bsz + b0 + arow) * Em, 0, 3);
#pragma unroll
      for (int kt = 0; kt < 4; ++kt) axn[kt] = ldA16(xr + kt * 32 + kbo);
    }

    v8f acc[2];
#pragma unroll
    for (int nt = 0; nt < 2; ++nt) {
      float bv = bias[nt];
#pragma unroll
      for (int r = 0; r < 8; ++r) acc[nt][r] = bv;
    }

#pragma unroll
    for (int kt = 0; kt < 4; ++kt) {
      v16h ah = ldA16(sH + arow * Hm + kt * 32 + kbo);
#pragma unroll
      for (int nt = 0; nt < 2; ++nt) {
        acc[nt] = __builtin_amdgcn_wmma_f32_16x16x32_f16(false, ax[kt], false, wih[nt][kt],
                                                         (short)0, acc[nt], false, false);
        acc[nt] = __builtin_amdgcn_wmma_f32_16x16x32_f16(false, ah, false, whh[nt][kt],
                                                         (short)0, acc[nt], false, false);
      }
    }

    // spill gate fragments so the cell can regroup i/f/g/o across waves
#pragma unroll
    for (int nt = 0; nt < 2; ++nt) {
      const int col = wv * 32 + nt * 16 + ncol;
#pragma unroll
      for (int r = 0; r < 8; ++r) sGate[(mrow + r) * G4 + col] = acc[nt][r];
    }
    __syncthreads();

    // fused LSTM cell: vectorized float4 gate reads, c persists in registers
    float4 gi4 = *(const float4*)&sGate[cm * G4 + cn];
    float4 gf4 = *(const float4*)&sGate[cm * G4 + 128 + cn];
    float4 gg4 = *(const float4*)&sGate[cm * G4 + 256 + cn];
    float4 go4 = *(const float4*)&sGate[cm * G4 + 384 + cn];
    float gi[4] = {gi4.x, gi4.y, gi4.z, gi4.w};
    float gf[4] = {gf4.x, gf4.y, gf4.z, gf4.w};
    float gg[4] = {gg4.x, gg4.y, gg4.z, gg4.w};
    float go[4] = {go4.x, go4.y, go4.z, go4.w};
    v4h hp;
#pragma unroll
    for (int r = 0; r < 4; ++r) {
      float cc = sigmoidf_(gf[r]) * c[r] + sigmoidf_(gi[r]) * tanhf(gg[r]);
      c[r] = cc;
      hp[r] = (f16)(sigmoidf_(go[r]) * tanhf(cc));
    }
    *(v4h*)&sH[cm * Hm + cn] = hp;
    *(v4h*)&h_all[(rowbase + cm) * HID2 + (size_t)dir * Hm + cn] = hp;
    __syncthreads();
  }
}

// ---------------- 4) emissions: (L*B,256) @ Wout^T (48,256) + bout ----------------
// Wout16 (24 KB) staged into LDS with async-to-LDS DMA (ASYNCcnt-tracked), then
// every wave reads its B fragments from LDS.
__global__ void k_emis(const f16* __restrict__ h_all, const f16* __restrict__ Wout16,
                       const float* __restrict__ bout, float* __restrict__ emis) {
  __shared__ f16 sW[Tt * HID2];              // 24 KB
  const int tid = threadIdx.x, lane = tid & 31, wv = tid >> 5;

  // async copy global -> LDS, 16B per transfer (CDNA5 ISA)
  {
    uint32_t lbase = (uint32_t)(uintptr_t)(&sW[0]);
#pragma unroll
    for (int i = tid; i < (Tt * HID2) / 8; i += 256) {
      uint32_t laddr = lbase + (uint32_t)i * 16u;
      uint64_t gaddr = (uint64_t)(uintptr_t)(Wout16 + (size_t)i * 8);
      asm volatile("global_load_async_to_lds_b128 %0, %1, off"
                   :: "v"(laddr), "v"(gaddr) : "memory");
    }
    asm volatile("s_wait_asynccnt 0x0" ::: "memory");
  }
  __syncthreads();

  const int mt = blockIdx.x * 8 + wv;        // M-tile 0..8191
  const int row0 = mt * 16;
  const int arow = lane & 15, ncol = lane & 15;
  const int koff = (lane >> 4) << 3;
  const int kbo  = (lane >> 4) << 4;

  v8f acc[3];
#pragma unroll
  for (int nt = 0; nt < 3; ++nt) {
    float bv = bout[nt * 16 + ncol];
#pragma unroll
    for (int r = 0; r < 8; ++r) acc[nt][r] = bv;
  }
  const f16* hrow = h_all + (size_t)(row0 + arow) * HID2;
#pragma unroll
  for (int kt = 0; kt < 8; ++kt) {
    const int k0 = kt * 32;
    v16h a = ldA16(hrow + k0 + kbo);
#pragma unroll
    for (int nt = 0; nt < 3; ++nt) {
      v16h b = *(const v16h*)(sW + (nt * 16 + ncol) * HID2 + k0 + kbo);
      acc[nt] = __builtin_amdgcn_wmma_f32_16x16x32_f16(false, a, false, b,
                                                       (short)0, acc[nt], false, false);
    }
  }
  const int mrow = (lane >> 4) * 8;
#pragma unroll
  for (int nt = 0; nt < 3; ++nt) {
    const int col = nt * 16 + ncol;
#pragma unroll
    for (int r = 0; r < 8; ++r)
      emis[(size_t)(row0 + mrow + r) * Tt + col] = acc[nt][r];
  }
}

// ---------------- 5) CRF numerator + forward algorithm, one block per batch row ----
__global__ void k_crf(const float* __restrict__ emis,   // (L*B, T)
                      const int* __restrict__ words,    // (B, L): mask == (word != 0)
                      const int* __restrict__ tags,     // (B, L)
                      const float* __restrict__ trans,  // (T, T)
                      const float* __restrict__ strans, const float* __restrict__ etrans,
                      float* __restrict__ partial) {    // (B): denom - num
  __shared__ float sTr[Tt * Tt];
  __shared__ float sSc[Tt];
  __shared__ float sEm[Tt];
  __shared__ float sNum;
  const int b = blockIdx.x;
  const int tid = threadIdx.x;             // 64 threads = 2 waves

  for (int i = tid; i < Tt * Tt; i += 64) sTr[i] = trans[i];
  if (tid < Tt) sSc[tid] = strans[tid] + emis[(size_t)b * Tt + tid];   // l = 0

  float num = 0.0f;
  int prevTag = 0, lastTag = 0;
  if (tid == 48) {
    int t0 = tags[(size_t)b * Lm];
    num = strans[t0] + emis[(size_t)b * Tt + t0];
    prevTag = t0; lastTag = t0;
  }

  for (int l = 1; l < Lm; ++l) {
    const int on = (words[(size_t)b * Lm + l] != 0);
    if (tid < Tt) sEm[tid] = emis[((size_t)l * Bsz + b) * Tt + tid];
    __syncthreads();
    float nxt = 0.0f;
    if (tid < Tt && on) {
      float mx = -1e30f;
      for (int i = 0; i < Tt; ++i) mx = fmaxf(mx, sSc[i] + sTr[i * Tt + tid]);
      float sum = 0.0f;
      for (int i = 0; i < Tt; ++i) sum += __expf(sSc[i] + sTr[i * Tt + tid] - mx);
      nxt = mx + logf(sum) + sEm[tid];
    }
    if (tid == 48 && on) {
      int tc = tags[(size_t)b * Lm + l];
      num += sTr[prevTag * Tt + tc] + sEm[tc];
      prevTag = tc; lastTag = tc;
    }
    __syncthreads();
    if (tid < Tt && on) sSc[tid] = nxt;
  }
  __syncthreads();
  if (tid == 48) sNum = num + etrans[lastTag];
  __syncthreads();
  if (tid == 0) {
    float mx = -1e30f;
    for (int j = 0; j < Tt; ++j) mx = fmaxf(mx, sSc[j] + etrans[j]);
    float sum = 0.0f;
    for (int j = 0; j < Tt; ++j) sum += __expf(sSc[j] + etrans[j] - mx);
    partial[b] = (mx + logf(sum)) - sNum;  // -llh_b
  }
}

// ---------------- 6) mean over batch ----------------
__global__ void k_reduce(const float* __restrict__ partial, float* __restrict__ out) {
  __shared__ float s[256];
  s[threadIdx.x] = partial[threadIdx.x];
  __syncthreads();
  for (int st = 128; st > 0; st >>= 1) {
    if (threadIdx.x < st) s[threadIdx.x] += s[threadIdx.x + st];
    __syncthreads();
  }
  if (threadIdx.x == 0) out[0] = s[0] / 256.0f;
}

// ---------------- host launch ----------------
extern "C" void kernel_launch(void* const* d_in, const int* in_sizes, int n_in,
                              void* d_out, int out_size, void* d_ws, size_t ws_size,
                              hipStream_t stream) {
  (void)in_sizes; (void)n_in; (void)out_size; (void)ws_size;
  const int*   words  = (const int*)d_in[0];
  const int*   tags   = (const int*)d_in[1];
  /* d_in[2] mask: derived from words!=0 instead */
  const float* emb    = (const float*)d_in[3];
  const float* Wih_f  = (const float*)d_in[4];
  const float* Whh_f  = (const float*)d_in[5];
  const float* bih_f  = (const float*)d_in[6];
  const float* bhh_f  = (const float*)d_in[7];
  const float* Wih_b  = (const float*)d_in[8];
  const float* Whh_b  = (const float*)d_in[9];
  const float* bih_b  = (const float*)d_in[10];
  const float* bhh_b  = (const float*)d_in[11];
  const float* Wout   = (const float*)d_in[12];
  const float* bout   = (const float*)d_in[13];
  const float* trans  = (const float*)d_in[14];
  const float* strans = (const float*)d_in[15];
  const float* etrans = (const float*)d_in[16];

  char* ws = (char*)d_ws;
  size_t o = 0;
  f16* x16 = (f16*)(ws + o);     o += (size_t)Lm * Bsz * Em * 2;    // 32 MB
  f16* h_all = (f16*)(ws + o);   o += (size_t)Lm * Bsz * HID2 * 2;  // 64 MB
  float* emis = (float*)(ws + o); o += (size_t)Lm * Bsz * Tt * 4;   // 24 MB
  f16* Wout16 = (f16*)(ws + o);  o += (size_t)Tt * HID2 * 2;
  float* partial = (float*)(ws + o); o += Bsz * 4;

  k_embed<<<4096, 256, 0, stream>>>(words, emb, x16);
  k_cvt_wout<<<1, 256, 0, stream>>>(Wout, Wout16);

  k_lstm<<<dim3(Bsz / 16, 2), 512, 0, stream>>>(x16,
      Wih_f, Whh_f, bih_f, bhh_f, Wih_b, Whh_b, bih_b, bhh_b, h_all);

  k_emis<<<(Lm * Bsz / 16) / 8, 256, 0, stream>>>(h_all, Wout16, bout, emis);
  k_crf<<<Bsz, 64, 0, stream>>>(emis, words, tags, trans, strans, etrans, partial);
  k_reduce<<<1, 256, 0, stream>>>(partial, (float*)d_out);
}
